// AccidentDetection_15066745274932
// MI455X (gfx1250) — matrix operand order repaired and checked by reading, not compiled, not run on hardware
//
#include <hip/hip_runtime.h>
#include <hip/hip_bf16.h>

// ---------------------------------------------------------------------------
// AccidentDetection forward for MI455X (gfx1250, wave32, WMMA).
//
// Math simplifications (exact up to FP reassociation):
//   * softmax over size-1 axis == 1  ->  alphas == obj_mask; attention dead.
//   * w_obj = ((sum_k x_k m_k^2) @ Wobj^T + b_obj*sum m^2) @ Wobj2^T + b_obj2*sum m
//
// v3: register double-buffered, software-pipelined WMMA inner loops
//   (loads for chunk k+32 issued before computing chunk k) in both the
//   feature GEMMs and the persistent-LSTM gate GEMM, so v_wmma no longer
//   sits behind an s_wait_loadcnt 0 on every fragment.
// ---------------------------------------------------------------------------

typedef __attribute__((ext_vector_type(16))) __bf16 v16bf;
typedef __attribute__((ext_vector_type(8)))  float  v8f;

namespace {

constexpr int  Bsz  = 32;
constexpr int  Nt   = 100;
constexpr int  Kobj = 19;     // K-1 objects
constexpr int  Din  = 4096;
constexpr int  DI   = 512;    // img/obj feature dim == H
constexpr int  Hd   = 512;

union BFrag { v16bf v; uint4 q[2]; };

// round-to-nearest-even f32 -> bf16 (packed pair / scalar)
__device__ __forceinline__ unsigned int pk2(float lo, float hi) {
  unsigned int a = __builtin_bit_cast(unsigned int, lo);
  unsigned int b = __builtin_bit_cast(unsigned int, hi);
  a += 0x7FFFu + ((a >> 16) & 1u);
  b += 0x7FFFu + ((b >> 16) & 1u);
  return (a >> 16) | (b & 0xFFFF0000u);
}
__device__ __forceinline__ unsigned short f2bf(float f) {
  unsigned int a = __builtin_bit_cast(unsigned int, f);
  a += 0x7FFFu + ((a >> 16) & 1u);
  return (unsigned short)(a >> 16);
}

// A fragment, 16x32, contiguous k order (not the ISA table order; B matches): lanes 0-15/16-31 both hold M=lane&15;
// VGPR 0-7 = K (lh*16 + 0..15), lh = lane>>4.
__device__ __forceinline__ void load_a_frag(BFrag& f, const unsigned short* __restrict__ a, int lh) {
  f.q[0] = *reinterpret_cast<const uint4*>(a + lh * 16);
  f.q[1] = *reinterpret_cast<const uint4*>(a + lh * 16 + 8);
}
// B fragment, 32x16: lane = N (lane&15); VGPR v = K pair (2v,2v+1) + 16*lh.
// w must already point at W[ncol*ldw + koff + lh*16]; 32 contiguous bytes.
__device__ __forceinline__ void load_b_frag(BFrag& f, const unsigned short* __restrict__ w) {
  f.q[0] = *reinterpret_cast<const uint4*>(w);
  f.q[1] = *reinterpret_cast<const uint4*>(w + 8);
}

#define WMMA_BF16(A, B, C) \
  __builtin_amdgcn_wmma_f32_16x16x32_bf16(false, (A), false, (B), (short)0, (C), false, false)

// ---------------------------------------------------------------------------
// Zero the LSTM state region (h/c buffers + barrier counter) each launch.
__global__ void zero_kernel(unsigned int* __restrict__ p, int n) {
  int i = blockIdx.x * 256 + threadIdx.x;
  if (i < n) p[i] = 0u;
}

// ---------------------------------------------------------------------------
// f32 -> packed bf16 streaming conversion (weights).
__global__ __launch_bounds__(256) void cvt_kernel(
    const float2* __restrict__ src, unsigned int* __restrict__ dst, int npairs)
{
  int i = blockIdx.x * 256 + threadIdx.x;
  if (i < npairs) { float2 v = src[i]; dst[i] = pk2(v.x, v.y); }
}

// ---------------------------------------------------------------------------
// Per (b,t): obj_mask -> alphas, s1 = sum m, s2 = sum m^2,
// Y = sum_k x_k m_k^2 (bf16), X0 = x[:, :, 0, :] (bf16).
__global__ __launch_bounds__(256) void masky_kernel(
    const float* __restrict__ x, float* __restrict__ alphas,
    unsigned int* __restrict__ Y, unsigned int* __restrict__ X0,
    float* __restrict__ s1, float* __restrict__ s2)
{
  const int r = blockIdx.x;           // r = b*Nt + t
  const int b = r / Nt, t = r % Nt;
  const float* xr = x + (long)r * 20 * Din;   // (b,t,0,:) base
  __shared__ float red[256];
  __shared__ float mk[Kobj], m2[Kobj];
  const int tid = threadIdx.x;

  for (int k = 0; k < Kobj; ++k) {
    const float* xo = xr + (long)(k + 1) * Din;
    float s = 0.0f;
    for (int d = tid; d < Din; d += 256) s += xo[d];
    red[tid] = s; __syncthreads();
    for (int st = 128; st > 0; st >>= 1) {
      if (tid < st) red[tid] += red[tid + st];
      __syncthreads();
    }
    if (tid == 0) { mk[k] = red[0]; m2[k] = red[0] * red[0]; }
    __syncthreads();
  }
  if (tid < Kobj) alphas[((long)t * Bsz + b) * Kobj + tid] = mk[tid];
  if (tid == 0) {
    float a = 0.0f, q = 0.0f;
    for (int k = 0; k < Kobj; ++k) { a += mk[k]; q += m2[k]; }
    s1[r] = a; s2[r] = q;
  }
  __syncthreads();

  unsigned int* Yd  = Y  + (long)r * (Din / 2);
  unsigned int* X0d = X0 + (long)r * (Din / 2);
  for (int i = tid; i < Din / 2; i += 256) {
    const int d = 2 * i;
    X0d[i] = pk2(xr[d], xr[d + 1]);
    float a0 = 0.0f, a1 = 0.0f;
    #pragma unroll
    for (int k = 0; k < Kobj; ++k) {
      const float* xo = xr + (long)(k + 1) * Din + d;
      a0 += xo[0] * m2[k];
      a1 += xo[1] * m2[k];
    }
    Yd[i] = pk2(a0, a1);
  }
}

// ---------------------------------------------------------------------------
// C[M x 512](bf16) = A(M x K bf16) @ W(512 x K bf16)^T
//                    + bias[col] * (scale ? scale[row] : 1).
// Grid (4, M/128); 8 waves (4m x 2n), each wave: 32 rows x 64 cols = 8 WMMA
// per 32-K chunk.  Double-buffered, peeled pipeline: fragments for chunk
// k+32 are in flight while chunk k is fed to the WMMA pipe.
__global__ __launch_bounds__(256) void gemm_bf16_nt(
    const unsigned short* __restrict__ A, long lda,
    const unsigned short* __restrict__ W, int K,
    unsigned short* __restrict__ C,
    const float* __restrict__ bias, const float* __restrict__ scale)
{
  const int tid = threadIdx.x;
  const int wv = tid >> 5, lane = tid & 31;
  const int l15 = lane & 15, lh = lane >> 4;
  const int wm = wv >> 1, wn = wv & 1;
  const int m0 = blockIdx.y * 128 + wm * 32;
  const int n0 = blockIdx.x * 128 + wn * 64;

  v8f acc[2][4] = {};
  const unsigned short* Ar0 = A + (long)(m0 + l15) * lda;
  const unsigned short* Ar1 = A + (long)(m0 + 16 + l15) * lda;
  const unsigned short* Wr[4];
  #pragma unroll
  for (int j = 0; j < 4; ++j)
    Wr[j] = W + (long)(n0 + j * 16 + l15) * K + lh * 16;

  BFrag a0A, a1A, bA[4];      // buffer A
  BFrag a0B, a1B, bB[4];      // buffer B

  auto ld = [&](BFrag& fa0, BFrag& fa1, BFrag (&fb)[4], int kc) {
    load_a_frag(fa0, Ar0 + kc, lh);
    load_a_frag(fa1, Ar1 + kc, lh);
    #pragma unroll
    for (int j = 0; j < 4; ++j) load_b_frag(fb[j], Wr[j] + kc);
  };
  auto comp = [&](BFrag& fa0, BFrag& fa1, BFrag (&fb)[4]) {
    #pragma unroll
    for (int j = 0; j < 4; ++j) {
      acc[0][j] = WMMA_BF16(fa0.v, fb[j].v, acc[0][j]);
      acc[1][j] = WMMA_BF16(fa1.v, fb[j].v, acc[1][j]);
    }
  };

  ld(a0A, a1A, bA, 0);
  int kc = 0;
  for (; kc + 64 < K; kc += 64) {
    ld(a0B, a1B, bB, kc + 32);
    __builtin_prefetch(Ar0 + kc + 512, 0, 3);
    __builtin_prefetch(Wr[0] + kc + 512, 0, 3);
    __builtin_prefetch(Wr[2] + kc + 512, 0, 3);
    comp(a0A, a1A, bA);
    ld(a0A, a1A, bA, kc + 64);
    comp(a0B, a1B, bB);
  }
  ld(a0B, a1B, bB, kc + 32);
  comp(a0A, a1A, bA);
  comp(a0B, a1B, bB);

  #pragma unroll
  for (int i = 0; i < 2; ++i) {
    #pragma unroll
    for (int j = 0; j < 4; ++j) {
      const int col = n0 + j * 16 + l15;
      const float bc = bias ? bias[col] : 0.0f;
      #pragma unroll
      for (int r = 0; r < 8; ++r) {
        const int row = m0 + i * 16 + lh * 8 + r;
        const float sv = scale ? scale[row] : 1.0f;
        C[(long)row * DI + col] = f2bf(acc[i][j][r] + bc * sv);
      }
    }
  }
}

// ---------------------------------------------------------------------------
// Persistent LSTM: one 512-long K segment of the gate GEMM (bf16 operands),
// same double-buffered pipeline as the feature GEMM.
__device__ __forceinline__ void gseg(
    v8f& a0, v8f& a1, const unsigned short* __restrict__ Arow,
    const unsigned short* __restrict__ W, int ldw, int koff,
    int nbase, int l15, int lh)
{
  const unsigned short* w0 = W + (long)(nbase + l15) * ldw + koff + lh * 16;
  const unsigned short* w1 = W + (long)(nbase + 16 + l15) * ldw + koff + lh * 16;

  BFrag aA, b0A, b1A, aB, b0B, b1B;
  auto ld = [&](BFrag& fa, BFrag& f0, BFrag& f1, int kc) {
    load_a_frag(fa, Arow + kc, lh);
    load_b_frag(f0, w0 + kc);
    load_b_frag(f1, w1 + kc);
  };

  ld(aA, b0A, b1A, 0);
  int kc = 0;
  for (; kc + 64 < 512; kc += 64) {
    ld(aB, b0B, b1B, kc + 32);
    a0 = WMMA_BF16(aA.v, b0A.v, a0);
    a1 = WMMA_BF16(aA.v, b1A.v, a1);
    ld(aA, b0A, b1A, kc + 64);
    a0 = WMMA_BF16(aB.v, b0B.v, a0);
    a1 = WMMA_BF16(aB.v, b1B.v, a1);
  }
  ld(aB, b0B, b1B, kc + 32);
  a0 = WMMA_BF16(aA.v, b0A.v, a0);
  a1 = WMMA_BF16(aA.v, b1A.v, a1);
  a0 = WMMA_BF16(aB.v, b0B.v, a0);
  a1 = WMMA_BF16(aB.v, b1B.v, a1);
}

__device__ __forceinline__ void activate(
    const float (*gl)[32][32],
    const float* __restrict__ bih, const float* __restrict__ bhh,
    float* __restrict__ Cb, unsigned int* __restrict__ Hc,
    float* __restrict__ hist, int hs, int tid)
{
  for (int idx = tid; idx < 512; idx += 256) {
    const int row = idx >> 4, cp = idx & 15;
    const int c0 = 2 * cp, c1 = c0 + 1;
    const int h0i = hs + c0, h1i = hs + c1;
    float hn[2], cn[2];
    #pragma unroll
    for (int e = 0; e < 2; ++e) {
      const int col = (e == 0) ? c0 : c1;
      const int hc  = (e == 0) ? h0i : h1i;
      const float gi = gl[0][row][col] + bih[hc]        + bhh[hc];
      const float gf = gl[1][row][col] + bih[512 + hc]  + bhh[512 + hc];
      const float gg = gl[2][row][col] + bih[1024 + hc] + bhh[1024 + hc];
      const float go = gl[3][row][col] + bih[1536 + hc] + bhh[1536 + hc];
      const float si = 1.0f / (1.0f + expf(-gi));
      const float sf = 1.0f / (1.0f + expf(-gf));
      const float so = 1.0f / (1.0f + expf(-go));
      cn[e] = sf * Cb[row * Hd + hc] + si * tanhf(gg);
      hn[e] = so * tanhf(cn[e]);
      Cb[row * Hd + hc] = cn[e];
      if (hist) hist[row * Hd + hc] = hn[e];
    }
    Hc[row * (Hd / 2) + (hs >> 1) + cp] = pk2(hn[0], hn[1]);
  }
}

// monotone global barrier across the 16 resident WGs
__device__ __forceinline__ void gsync(unsigned int* cnt, unsigned int* gen,
                                      int nwg, int tid)
{
  __threadfence();
  __syncthreads();
  if (tid == 0) {
    atomicAdd(cnt, 1u);
    *gen += (unsigned)nwg;
    volatile unsigned int* vc = cnt;
    while (*vc < *gen) __builtin_amdgcn_s_sleep(2);
  }
  __syncthreads();
  __threadfence();
}

// 16 WGs; WG blk owns hidden cols [32*blk, 32*blk+32) for both layers:
// all 4 gates for those columns live in this WG -> activations are local.
// 8 waves: wave = gate*2 + batch-half; per wave 2 WMMA accumulators.
__global__ __launch_bounds__(256) void lstm_kernel(
    const unsigned short* __restrict__ IMG, const unsigned short* __restrict__ WOB,
    const unsigned short* __restrict__ Wih0, const unsigned short* __restrict__ Whh0,
    const float* __restrict__ bih0, const float* __restrict__ bhh0,
    const unsigned short* __restrict__ Wih1, const unsigned short* __restrict__ Whh1,
    const float* __restrict__ bih1, const float* __restrict__ bhh1,
    unsigned short* __restrict__ H0, unsigned short* __restrict__ H1,
    float* __restrict__ C0, float* __restrict__ C1,
    float* __restrict__ HIST, unsigned int* __restrict__ cnt)
{
  __shared__ float gl[4][32][32];          // 16 KB gate staging
  const int tid   = threadIdx.x;
  const int wave  = tid >> 5, lane = tid & 31;
  const int l15   = lane & 15, lh = lane >> 4;
  const int g     = wave >> 1, mhalf = wave & 1;
  const int hs    = (int)blockIdx.x * 32;
  const int nbase = g * 512 + hs;
  const int mrow  = mhalf * 16 + l15;      // batch row this lane loads for A
  const long bstr = (long)Nt * DI;         // row stride of IMG/WOB (b-major)
  const int  nwg  = (int)gridDim.x;
  unsigned int gen = 0;

  for (int t = 0; t < Nt; ++t) {
    const unsigned short* h0p = H0 + (t & 1) * (Bsz * Hd);
    unsigned short*       h0c = H0 + ((t + 1) & 1) * (Bsz * Hd);
    const unsigned short* h1p = H1 + (t & 1) * (Bsz * Hd);
    unsigned short*       h1c = H1 + ((t + 1) & 1) * (Bsz * Hd);

    // layer 0: gates = [img|wobj]@Wih0^T + h0@Whh0^T
    v8f a0 = {}, a1 = {};
    gseg(a0, a1, IMG + (long)t * DI + (long)mrow * bstr, Wih0, 2 * DI, 0,  nbase, l15, lh);
    gseg(a0, a1, WOB + (long)t * DI + (long)mrow * bstr, Wih0, 2 * DI, DI, nbase, l15, lh);
    gseg(a0, a1, h0p + mrow * Hd,                        Whh0, Hd,     0,  nbase, l15, lh);
    #pragma unroll
    for (int r = 0; r < 8; ++r) {
      const int row = mhalf * 16 + lh * 8 + r;
      gl[g][row][l15]      = a0[r];
      gl[g][row][16 + l15] = a1[r];
    }
    __syncthreads();
    activate(gl, bih0, bhh0, C0, (unsigned int*)h0c, nullptr, hs, tid);
    gsync(cnt, &gen, nwg, tid);

    // layer 1: gates = h0_new@Wih1^T + h1@Whh1^T
    v8f b0 = {}, b1 = {};
    gseg(b0, b1, h0c + mrow * Hd, Wih1, Hd, 0, nbase, l15, lh);
    gseg(b0, b1, h1p + mrow * Hd, Whh1, Hd, 0, nbase, l15, lh);
    #pragma unroll
    for (int r = 0; r < 8; ++r) {
      const int row = mhalf * 16 + lh * 8 + r;
      gl[g][row][l15]      = b0[r];
      gl[g][row][16 + l15] = b1[r];
    }
    __syncthreads();
    activate(gl, bih1, bhh1, C1, (unsigned int*)h1c, HIST + (long)t * Bsz * Hd, hs, tid);
    gsync(cnt, &gen, nwg, tid);
  }
}

// ---------------------------------------------------------------------------
// preds[b,t] = H1hist[t,b,:] . Wout + b_out
__global__ __launch_bounds__(256) void pred_kernel(
    const float* __restrict__ HIST, const float* __restrict__ Wout,
    const float* __restrict__ bout, float* __restrict__ out)
{
  const int r = blockIdx.x;           // r = b*Nt + t
  const int b = r / Nt, t = r % Nt;
  __shared__ float red[256];
  const float* h = HIST + ((long)t * Bsz + b) * Hd;
  float s = 0.0f;
  for (int j = threadIdx.x; j < Hd; j += 256) s += h[j] * Wout[j];
  red[threadIdx.x] = s; __syncthreads();
  for (int st = 128; st > 0; st >>= 1) {
    if (threadIdx.x < st) red[threadIdx.x] += red[threadIdx.x + st];
    __syncthreads();
  }
  if (threadIdx.x == 0) out[b * Nt + t] = red[0] + bout[0];
}

} // namespace

// ---------------------------------------------------------------------------
extern "C" void kernel_launch(void* const* d_in, const int* in_sizes, int n_in,
                              void* d_out, int out_size, void* d_ws, size_t ws_size,
                              hipStream_t stream)
{
  (void)in_sizes; (void)n_in; (void)out_size; (void)ws_size;

  const float* x     = (const float*)d_in[0];
  const float* Wimg  = (const float*)d_in[1];
  const float* bimg  = (const float*)d_in[2];
  const float* Wobj  = (const float*)d_in[3];
  const float* bobj  = (const float*)d_in[4];
  const float* Wobj2 = (const float*)d_in[5];
  const float* bobj2 = (const float*)d_in[6];
  // d_in[7] (W_ah) and d_in[8] (W_ac) are dead: softmax over size-1 axis == 1.
  const float* Wih0 = (const float*)d_in[9];
  const float* Whh0 = (const float*)d_in[10];
  const float* bih0 = (const float*)d_in[11];
  const float* bhh0 = (const float*)d_in[12];
  const float* Wih1 = (const float*)d_in[13];
  const float* Whh1 = (const float*)d_in[14];
  const float* bih1 = (const float*)d_in[15];
  const float* bhh1 = (const float*)d_in[16];
  const float* Wout = (const float*)d_in[17];
  const float* bout = (const float*)d_in[18];

  float* alphas = (float*)d_out;                       // (N,B,19,1) = 60800
  float* preds  = (float*)d_out + (long)Nt * Bsz * Kobj;

  // workspace layout (bytes, all offsets 256B-aligned)
  char* wsb = (char*)d_ws;
  unsigned short* Ybf    = (unsigned short*)(wsb + 0L);          // 3200x4096 bf16
  unsigned short* X0bf   = (unsigned short*)(wsb + 26214400L);   // 3200x4096 bf16
  unsigned short* IMGbf  = (unsigned short*)(wsb + 52428800L);   // 3200x512
  unsigned short* WOBbf  = (unsigned short*)(wsb + 55705600L);   // 3200x512
  unsigned short* Ubf    = (unsigned short*)(wsb + 58982400L);   // 3200x512
  unsigned short* Wimgb  = (unsigned short*)(wsb + 62259200L);   // 512x4096
  unsigned short* Wobjb  = (unsigned short*)(wsb + 66453504L);   // 512x4096
  unsigned short* Wobj2b = (unsigned short*)(wsb + 70647808L);   // 512x512
  unsigned short* Wih0b  = (unsigned short*)(wsb + 71172096L);   // 2048x1024
  unsigned short* Whh0b  = (unsigned short*)(wsb + 75366400L);   // 2048x512
  unsigned short* Wih1b  = (unsigned short*)(wsb + 77463552L);   // 2048x512
  unsigned short* Whh1b  = (unsigned short*)(wsb + 79560704L);   // 2048x512
  float*          HIST   = (float*)(wsb + 81657856L);            // 100x32x512 f32
  float*          s1     = (float*)(wsb + 88211456L);            // 3200
  float*          s2     = (float*)(wsb + 88224256L);            // 3200
  unsigned short* H0     = (unsigned short*)(wsb + 88237056L);   // 2x32x512 bf16
  unsigned short* H1     = (unsigned short*)(wsb + 88302592L);   // 2x32x512 bf16
  float*          C0     = (float*)(wsb + 88368128L);            // 32x512 f32
  float*          C1     = (float*)(wsb + 88433664L);            // 32x512 f32
  unsigned int*   cnt    = (unsigned int*)(wsb + 88499200L);

  // 1) deterministic state init (h/c/cnt region = 262400 bytes = 65600 dwords)
  zero_kernel<<<257, 256, 0, stream>>>((unsigned int*)(wsb + 88237056L), 65600);

  // 2) mask / alphas / s1 / s2 / Y(bf16) / X0(bf16)
  masky_kernel<<<Bsz * Nt, 256, 0, stream>>>(x, alphas,
      (unsigned int*)Ybf, (unsigned int*)X0bf, s1, s2);

  // 3) one-time weight conversions f32 -> bf16
  cvt_kernel<<<4096, 256, 0, stream>>>((const float2*)Wimg,  (unsigned int*)Wimgb,  1048576);
  cvt_kernel<<<4096, 256, 0, stream>>>((const float2*)Wobj,  (unsigned int*)Wobjb,  1048576);
  cvt_kernel<<< 512, 256, 0, stream>>>((const float2*)Wobj2, (unsigned int*)Wobj2b,  131072);
  cvt_kernel<<<4096, 256, 0, stream>>>((const float2*)Wih0,  (unsigned int*)Wih0b,  1048576);
  cvt_kernel<<<2048, 256, 0, stream>>>((const float2*)Whh0,  (unsigned int*)Whh0b,   524288);
  cvt_kernel<<<2048, 256, 0, stream>>>((const float2*)Wih1,  (unsigned int*)Wih1b,   524288);
  cvt_kernel<<<2048, 256, 0, stream>>>((const float2*)Whh1,  (unsigned int*)Whh1b,   524288);

  // 4) feature GEMMs (M=3200, N=512), bf16 WMMA
  gemm_bf16_nt<<<dim3(4, 25), 256, 0, stream>>>(X0bf, (long)Din, Wimgb,  Din, IMGbf, bimg,  nullptr);
  gemm_bf16_nt<<<dim3(4, 25), 256, 0, stream>>>(Ybf,  (long)Din, Wobjb,  Din, Ubf,   bobj,  s2);
  gemm_bf16_nt<<<dim3(4, 25), 256, 0, stream>>>(Ubf,  (long)DI,  Wobj2b, DI,  WOBbf, bobj2, s1);

  // 5) persistent two-layer LSTM over all 100 steps (16 WGs + global barrier)
  lstm_kernel<<<16, 256, 0, stream>>>(IMGbf, WOBbf, Wih0b, Whh0b, bih0, bhh0,
                                      Wih1b, Whh1b, bih1, bhh1,
                                      H0, H1, C0, C1, HIST, cnt);

  // 6) output head
  pred_kernel<<<Bsz * Nt, 256, 0, stream>>>(HIST, Wout, bout, preds);
}
